// LearnableCodebook_58841051955467
// MI455X (gfx1250) — matrix operand order repaired and checked by reading, not compiled, not run on hardware
//
#include <hip/hip_runtime.h>

// ---------------------------------------------------------------------------
// LearnableCodebook eval pass for MI455X (gfx1250, wave32, WMMA).
//   cluster_centers, assignments = gather(protos, argmax_k cos(x, protos))
// B=32, N=4096, D=128, K=512. Similarity GEMM done in f16 WMMA with fp32
// accumulation; argmax fused in-register (the [B,N,K] tensor never exists).
// ---------------------------------------------------------------------------

typedef __attribute__((ext_vector_type(16))) _Float16 v16h;
typedef __attribute__((ext_vector_type(8)))  float    v8f;
typedef __attribute__((ext_vector_type(4)))  _Float16 h4;

namespace {
constexpr int Bb = 32, Nn = 4096, Dd = 128, Kk = 512;
constexpr int M_TOTAL  = Bb * Nn;       // 131072 rows
constexpr int TILE_M   = 128;           // rows per workgroup (8 waves x 16)
constexpr int CHUNK    = 128;           // prototypes staged in LDS per pass
constexpr int LDS_PAD  = 8;             // f16 elems: row stride 272B -> bank-conflict free
constexpr int RSTRIDE  = Dd + LDS_PAD;  // 136 f16
constexpr float EPS    = 1e-12f;
}

// ---- Kernel 1: L2-normalize prototypes, store as f16 (one wave per row) ----
__global__ __launch_bounds__(256) void proto_norm_f16_kernel(
    const float* __restrict__ protos, _Float16* __restrict__ protos_h) {
  const int row  = blockIdx.x * 8 + (threadIdx.x >> 5);   // 8 waves/block
  const int lane = threadIdx.x & 31;
  if (row >= Kk) return;
  const float4 v = reinterpret_cast<const float4*>(protos + (size_t)row * Dd)[lane];
  float ss = v.x * v.x + v.y * v.y + v.z * v.z + v.w * v.w;
#pragma unroll
  for (int off = 16; off; off >>= 1) ss += __shfl_xor(ss, off, 32);
  const float s = 1.0f / fmaxf(sqrtf(ss), EPS);
  h4 o = { (_Float16)(v.x * s), (_Float16)(v.y * s),
           (_Float16)(v.z * s), (_Float16)(v.w * s) };
  *reinterpret_cast<h4*>(protos_h + (size_t)row * Dd + lane * 4) = o;
}

// ---- Kernel 2: normalize x tile -> WMMA similarity -> fused argmax -> gather
__global__ __launch_bounds__(256) void codebook_assign_kernel(
    const float* __restrict__ x, const float* __restrict__ protos,
    const _Float16* __restrict__ protos_h,
    float* __restrict__ out_centers, float* __restrict__ out_assign) {

  __shared__ _Float16 xs[TILE_M][RSTRIDE];   // normalized x tile (f16)   ~34.8 KB
  __shared__ _Float16 ps[CHUNK][RSTRIDE];    // normalized proto chunk    ~34.8 KB

  const int tid    = threadIdx.x;
  const int wave   = tid >> 5;
  const int lane   = tid & 31;
  const int half   = lane >> 4;      // lane-half selects K sub-block in fragments
  const int l16    = lane & 15;
  const int m_base = blockIdx.x * TILE_M;
  const int wrow   = wave * 16;      // this wave's 16-row strip inside the tile

  // ---- Stage 1: stream-load + L2-normalize 16 rows per wave into LDS (f16)
  for (int r = 0; r < 16; ++r) {
    const int row = wrow + r;
    const float4 v =
        reinterpret_cast<const float4*>(x + (size_t)(m_base + row) * Dd)[lane];
    float ss = v.x * v.x + v.y * v.y + v.z * v.z + v.w * v.w;
#pragma unroll
    for (int off = 16; off; off >>= 1) ss += __shfl_xor(ss, off, 32);
    const float s = 1.0f / fmaxf(sqrtf(ss), EPS);
    h4 o = { (_Float16)(v.x * s), (_Float16)(v.y * s),
             (_Float16)(v.z * s), (_Float16)(v.w * s) };
    *reinterpret_cast<h4*>(&xs[row][lane * 4]) = o;
  }
  __syncthreads();

  // ---- Preload the 4 A-fragments (K=0..127) for this wave's 16-row strip.
  // ISA 16-bit A 16x32 layout: lane(0-15)=M; V0..3:K=k0+8*half+0..7,
  //                            V4..7:K=k0+16+8*half+0..7.
  v16h afrag[4];
#pragma unroll
  for (int ks = 0; ks < 4; ++ks) {
    const int k0 = ks * 32;
    const _Float16* rp = &xs[wrow + l16][0];
    union { v16h v; uint4 q[2]; } f;
    f.q[0] = *reinterpret_cast<const uint4*>(rp + k0 + 8 * half);
    f.q[1] = *reinterpret_cast<const uint4*>(rp + k0 + 16 + 8 * half);
    afrag[ks] = f.v;
  }

  // Running per-lane argmax over the column slots this lane observes.
  float bestv[8];
  int   besti[8];
#pragma unroll
  for (int i = 0; i < 8; ++i) { bestv[i] = -3.0e38f; besti[i] = 0x7fffffff; }

  for (int chunk = 0; chunk < 4; ++chunk) {
    // Stage 128 prototype rows (f16) into LDS; cooperative B128 copies.
    {
      const _Float16* src = protos_h + (size_t)chunk * CHUNK * Dd;
      for (int i = tid; i < CHUNK * (Dd / 8); i += 256) {   // 1 item = 8 f16
        const int rrow = i >> 4;           // 16 uint4 per row
        const int rcol = (i & 15) * 8;
        *reinterpret_cast<uint4*>(&ps[rrow][rcol]) =
            *reinterpret_cast<const uint4*>(src + (size_t)rrow * Dd + rcol);
      }
      if (chunk < 3) {   // pull next chunk toward the caches (global_prefetch_b8)
        const char* nxt =
            reinterpret_cast<const char*>(protos_h + (size_t)(chunk + 1) * CHUNK * Dd);
        __builtin_prefetch(nxt + tid * 128, 0, 0);
      }
    }
    __syncthreads();

#pragma unroll
    for (int t = 0; t < 8; ++t) {          // 8 N-tiles of 16 prototypes
      v8f acc = {};
#pragma unroll
      for (int ks = 0; ks < 4; ++ks) {     // K=128 in 4 WMMA steps
        const int k0 = ks * 32;
        // B 32x16 layout (matches A): lane(0-15)=N; V0..3:K=k0+8*half+0..7,
        // V4..7:K=k0+16+8*half+0..7.
        const _Float16* rp = &ps[t * 16 + l16][0];
        union { v16h v; uint4 q[2]; } f;
        f.q[0] = *reinterpret_cast<const uint4*>(rp + k0 + 8 * half);
        f.q[1] = *reinterpret_cast<const uint4*>(rp + k0 + 16 + 8 * half);
        acc = __builtin_amdgcn_wmma_f32_16x16x32_f16(
            false, afrag[ks], false, f.v, (short)0, acc, false, false);
      }
      // C layout: VGPR i, lane -> (M = i + 8*half, N = l16). Each lane tracks
      // its own column slot; col strictly increases, so '>' keeps lowest tie.
      const int col = chunk * 128 + t * 16 + l16;
#pragma unroll
      for (int i = 0; i < 8; ++i) {
        if (acc[i] > bestv[i]) { bestv[i] = acc[i]; besti[i] = col; }
      }
    }
    __syncthreads();   // done with this ps chunk before restaging
  }

  // ---- Cross-lane argmax: xor-butterfly within each 16-lane half (N axis).
#pragma unroll
  for (int i = 0; i < 8; ++i) {
#pragma unroll
    for (int off = 8; off; off >>= 1) {
      const float ov = __shfl_xor(bestv[i], off, 32);
      const int   oi = __shfl_xor(besti[i], off, 32);
      if (ov > bestv[i] || (ov == bestv[i] && oi < besti[i])) {
        bestv[i] = ov; besti[i] = oi;
      }
    }
  }
  // Now every lane in a half holds the argmax for rows (i + 8*half).

  if (l16 == 0) {      // lane 0 -> rows 0..7, lane 16 -> rows 8..15
#pragma unroll
    for (int i = 0; i < 8; ++i) {
      out_assign[m_base + wrow + half * 8 + i] = (float)besti[i];
    }
  }

  // ---- Gather un-normalized prototype rows -> cluster centers (float4 copies)
#pragma unroll
  for (int r = 0; r < 16; ++r) {
    const int src_lane = (r < 8) ? 0 : 16;
    const int k   = __shfl(besti[r & 7], src_lane, 32);
    const int row = m_base + wrow + r;
    reinterpret_cast<float4*>(out_centers + (size_t)row * Dd)[lane] =
        reinterpret_cast<const float4*>(protos + (size_t)k * Dd)[lane];
  }
}

extern "C" void kernel_launch(void* const* d_in, const int* in_sizes, int n_in,
                              void* d_out, int out_size, void* d_ws, size_t ws_size,
                              hipStream_t stream) {
  (void)in_sizes; (void)n_in; (void)out_size; (void)ws_size;
  const float* x      = (const float*)d_in[0];   // (B,N,D) fp32
  const float* protos = (const float*)d_in[1];   // (K,D)   fp32
  _Float16* protos_h  = (_Float16*)d_ws;         // 512*128*2 = 128 KB scratch

  float* out_centers = (float*)d_out;                          // (B,N,D)
  float* out_assign  = out_centers + (size_t)M_TOTAL * Dd;     // (B,N) as float

  proto_norm_f16_kernel<<<Kk / 8, 256, 0, stream>>>(protos, protos_h);
  codebook_assign_kernel<<<M_TOTAL / TILE_M, 256, 0, stream>>>(
      x, protos, protos_h, out_centers, out_assign);
}
